// DeltaNetMolecular_36438502540092
// MI455X (gfx1250) — matrix operand order repaired and checked by reading, not compiled, not run on hardware
//
#include <hip/hip_runtime.h>
#include <hip/hip_bf16.h>
#include <cstdint>

// ---------------------------------------------------------------------------
// Types for gfx1250 WMMA (wave32): v_wmma_f32_16x16x32_bf16
// ---------------------------------------------------------------------------
typedef __bf16 bf16;
typedef bf16  v16bf __attribute__((ext_vector_type(16)));
typedef bf16  v8bf  __attribute__((ext_vector_type(8)));
typedef float v8f   __attribute__((ext_vector_type(8)));

__device__ __forceinline__ v8f wmma_bf16(v16bf a, v16bf b, v8f c) {
  // (neg_a, A, neg_b, B, c_mod, C, reuse_a, reuse_b)
  return __builtin_amdgcn_wmma_f32_16x16x32_bf16(false, a, false, b, (short)0, c,
                                                 false, false);
}

__device__ __forceinline__ float silu(float x) { return x / (1.0f + __expf(-x)); }

// A-fragment (16x32 bf16, MxK): lane = (m, half). Per-lane two contiguous
// 8-elem chunks at K = half*8 and K = 16 + half*8 (ISA 7.12.2).
__device__ __forceinline__ v16bf load_a(const bf16* row, int kt, int half) {
  v8bf c0 = *(const v8bf*)(row + kt * 32 + half * 8);
  v8bf c1 = *(const v8bf*)(row + kt * 32 + 16 + half * 8);
  v16bf a;
#pragma unroll
  for (int i = 0; i < 8; ++i) { a[i] = c0[i]; a[i + 8] = c1[i]; }
  return a;
}

// B-fragment (32x16 bf16, KxN): lane (n=m, half) holds two 8-elem chunks at
// K = half*8 and K = 16 + half*8, from packed Wt laid out [n][Kpad] (K contiguous).
__device__ __forceinline__ v16bf load_b(const bf16* col, int kt, int half) {
  v8bf c0 = *(const v8bf*)(col + kt * 32 + half * 8);
  v8bf c1 = *(const v8bf*)(col + kt * 32 + 16 + half * 8);
  v16bf b;
#pragma unroll
  for (int i = 0; i < 8; ++i) { b[i] = c0[i]; b[i + 8] = c1[i]; }
  return b;
}

// ---------------------------------------------------------------------------
// Weight packing: fp32 (din,dout) row-major  ->  bf16 Wt (Npad, Kpad), zeros
// beyond (dout, din).  Wt[n][k] = w[k][n].
// ---------------------------------------------------------------------------
__global__ void pack_wt(const float* __restrict__ w, bf16* __restrict__ wt,
                        int din, int dout, int Kpad, int Npad) {
  int idx = blockIdx.x * 256 + threadIdx.x;
  if (idx >= Npad * Kpad) return;
  int n = idx / Kpad, k = idx % Kpad;
  float v = (k < din && n < dout) ? w[(size_t)k * dout + n] : 0.0f;
  wt[idx] = (bf16)v;
}

// ---------------------------------------------------------------------------
// Generic wave-tiled GEMM: out = [silu](A[M,Kpad]@W + b) [+ resid], 16 rows
// per wave, 4 waves per block.  A is bf16 row-major, Kpad % 32 == 0,
// Nout % 16 == 0.  Rows padded to multiple of 64 in A; stores guarded by M.
// ---------------------------------------------------------------------------
__global__ __launch_bounds__(128) void gemm16(
    const bf16* __restrict__ A, const bf16* __restrict__ Wt,
    const float* __restrict__ bias, int M, int Kpad, int Nout,
    float* __restrict__ outF, bf16* __restrict__ outB,
    const float* __restrict__ resid, int doSilu) {
  const int tid = threadIdx.x;
  const int wave = tid >> 5, lane = tid & 31;
  const int m = lane & 15, half = lane >> 4;
  const int rb = (blockIdx.x * 4 + wave) * 16;
  const bf16* arow = A + (size_t)(rb + m) * Kpad;
  const int nkt = Kpad >> 5;
  for (int nt = 0; nt * 16 < Nout; ++nt) {
    int col = nt * 16 + m;
    float bv = bias ? bias[col] : 0.0f;
    v8f acc = {bv, bv, bv, bv, bv, bv, bv, bv};
    const bf16* wcol = Wt + (size_t)col * Kpad;
    for (int kt = 0; kt < nkt; ++kt)
      acc = wmma_bf16(load_a(arow, kt, half), load_b(wcol, kt, half), acc);
#pragma unroll
    for (int r = 0; r < 8; ++r) {
      int row = rb + r + 8 * half;
      if (row < M) {
        float x = acc[r];
        if (doSilu) x = silu(x);
        if (resid) x += resid[(size_t)row * Nout + col];
        if (outF) outF[(size_t)row * Nout + col] = x;
        if (outB) outB[(size_t)row * Nout + col] = (bf16)x;
      }
    }
  }
}

// ---------------------------------------------------------------------------
// Fused per-edge kernel: builds edge_in (16x288 bf16) in LDS, runs
// edge1(265->530)+silu streamed straight into edge2(530->64)+silu (hidden
// never leaves the WGP), then the coors MLP (64->256->1), and atomically
// scatters m and cw*rel into per-dst accumulators.
// ---------------------------------------------------------------------------
__global__ __launch_bounds__(128) void edge_kernel(
    const float* __restrict__ coords, const float* __restrict__ feats,
    const int* __restrict__ srcIdx, const int* __restrict__ dstIdx,
    const bf16* __restrict__ wt1, const float* __restrict__ b1,   // 288x544
    const bf16* __restrict__ wt2, const float* __restrict__ b2,   // 544x64
    const bf16* __restrict__ wtc1, const float* __restrict__ bc1, // 64x256
    const float* __restrict__ wc2, const float* __restrict__ bc2, // 256 , 1
    float* __restrict__ m_sum, float* __restrict__ cwrel_sum, int E) {
  constexpr int KP1 = 288;  // edge_in K padded (265 -> 288)
  constexpr int H1 = 544;   // edge hidden padded (530 -> 544)
  __shared__ bf16 Xin[4][16][KP1];
  __shared__ bf16 stage[4][16][32];
  __shared__ float relS[4][16][3];
  __shared__ int dstS[4][16];

  const int tid = threadIdx.x;
  const int wave = tid >> 5, lane = tid & 31;
  const int m = lane & 15, half = lane >> 4;
  const int eb = (blockIdx.x * 4 + wave) * 16;

  // ---- build edge-input tile --------------------------------------------
  {
    int e = eb + m;
    int s = 0, d = 0;
    if (e < E) { s = srcIdx[e]; d = dstIdx[e]; }
    if (half == 0) {
      dstS[wave][m] = (e < E) ? d : -1;
      float rx = 0.f, ry = 0.f, rz = 0.f;
      if (e < E) {
        rx = coords[s * 3 + 0] - coords[d * 3 + 0];
        ry = coords[s * 3 + 1] - coords[d * 3 + 1];
        rz = coords[s * 3 + 2] - coords[d * 3 + 2];
      }
      relS[wave][m][0] = rx; relS[wave][m][1] = ry; relS[wave][m][2] = rz;
      float d2 = rx * rx + ry * ry + rz * rz;
#pragma unroll
      for (int f = 0; f < 4; ++f) {
        float xf = d2 / (float)(1 << f);
        Xin[wave][m][256 + f] = (bf16)__sinf(xf);
        Xin[wave][m][260 + f] = (bf16)__cosf(xf);
      }
      Xin[wave][m][264] = (bf16)d2;
#pragma unroll
      for (int k = 265; k < KP1; ++k) Xin[wave][m][k] = (bf16)0.0f;
    }
    // gather feats: half 0 -> feats[dst] (cols 0..127), half 1 -> feats[src]
    int node = half ? s : d;
    const float4* fp = (const float4*)(feats + (size_t)node * 128);
#pragma unroll 8
    for (int i = 0; i < 32; ++i) {
      float4 v = fp[i];
      int c = half * 128 + i * 4;
      Xin[wave][m][c + 0] = (bf16)v.x;
      Xin[wave][m][c + 1] = (bf16)v.y;
      Xin[wave][m][c + 2] = (bf16)v.z;
      Xin[wave][m][c + 3] = (bf16)v.w;
    }
  }
  asm volatile("s_wait_dscnt 0x0" ::: "memory");  // wave-local LDS stage ready

  // ---- edge1 -> silu streamed into edge2 accumulation --------------------
  v8f macc[4];
#pragma unroll
  for (int t = 0; t < 4; ++t) {
    float bv = b2[t * 16 + m];
#pragma unroll
    for (int r = 0; r < 8; ++r) macc[t][r] = bv;
  }
  const bf16* arow = &Xin[wave][m][0];
  for (int jp = 0; jp < H1 / 32; ++jp) {   // 17 pairs of hidden tiles
#pragma unroll
    for (int jh = 0; jh < 2; ++jh) {
      int col = (jp * 2 + jh) * 16 + m;
      float bv = (col < 530) ? b1[col] : 0.0f;
      v8f acc = {bv, bv, bv, bv, bv, bv, bv, bv};
      const bf16* wcol = wt1 + (size_t)col * KP1;
#pragma unroll
      for (int kt = 0; kt < KP1 / 32; ++kt)
        acc = wmma_bf16(load_a(arow, kt, half), load_b(wcol, kt, half), acc);
#pragma unroll
      for (int r = 0; r < 8; ++r)
        stage[wave][r + 8 * half][jh * 16 + m] = (bf16)silu(acc[r]);
    }
    asm volatile("s_wait_dscnt 0x0" ::: "memory");
    v16bf ah = load_a(&stage[wave][m][0], 0, half);  // hidden as K-slice
#pragma unroll
    for (int t = 0; t < 4; ++t) {
      const bf16* wcol = wt2 + (size_t)(t * 16 + m) * H1 + jp * 32;
      macc[t] = wmma_bf16(ah, load_b(wcol, 0, half), macc[t]);
    }
  }
  // edge2 silu -> m, scatter into m_sum[dst]
#pragma unroll
  for (int t = 0; t < 4; ++t)
#pragma unroll
    for (int r = 0; r < 8; ++r) macc[t][r] = silu(macc[t][r]);
#pragma unroll
  for (int t = 0; t < 4; ++t)
#pragma unroll
    for (int r = 0; r < 8; ++r) {
      int d = dstS[wave][r + 8 * half];
      if (d >= 0) atomicAdd(&m_sum[(size_t)d * 64 + t * 16 + m], macc[t][r]);
    }

  // ---- coors MLP: m(16x64) -> 256 (silu) -> dot(wc2) ---------------------
  v16bf A01[2];
#pragma unroll
  for (int p = 0; p < 2; ++p) {
#pragma unroll
    for (int t = 0; t < 2; ++t)
#pragma unroll
      for (int r = 0; r < 8; ++r)
        stage[wave][r + 8 * half][t * 16 + m] = (bf16)macc[p * 2 + t][r];
    asm volatile("s_wait_dscnt 0x0" ::: "memory");
    A01[p] = load_a(&stage[wave][m][0], 0, half);
  }
  float p8[8];
#pragma unroll
  for (int r = 0; r < 8; ++r) p8[r] = 0.0f;
  for (int nt = 0; nt < 16; ++nt) {
    float bv = bc1[nt * 16 + m];
    v8f acc = {bv, bv, bv, bv, bv, bv, bv, bv};
    const bf16* wcol = wtc1 + (size_t)(nt * 16 + m) * 64;
    acc = wmma_bf16(A01[0], load_b(wcol, 0, half), acc);
    acc = wmma_bf16(A01[1], load_b(wcol, 1, half), acc);
    float w2 = wc2[nt * 16 + m];
#pragma unroll
    for (int r = 0; r < 8; ++r) p8[r] += silu(acc[r]) * w2;
  }
#pragma unroll
  for (int off = 1; off < 16; off <<= 1)
#pragma unroll
    for (int r = 0; r < 8; ++r) p8[r] += __shfl_xor(p8[r], off, 32);
  float cb = bc2[0];
  if (m < 8) {
    int row = 8 * half + m;
    int d = dstS[wave][row];
    if (d >= 0) {
      float cw = p8[m] + cb;
      atomicAdd(&cwrel_sum[(size_t)d * 3 + 0], cw * relS[wave][row][0]);
      atomicAdd(&cwrel_sum[(size_t)d * 3 + 1], cw * relS[wave][row][1]);
      atomicAdd(&cwrel_sum[(size_t)d * 3 + 2], cw * relS[wave][row][2]);
    }
  }
}

// ---------------------------------------------------------------------------
// Elementwise / assembly kernels
// ---------------------------------------------------------------------------
__global__ void deg_kernel(const int* __restrict__ dst, float* __restrict__ deg,
                           int E) {
  int e = blockIdx.x * 256 + threadIdx.x;
  if (e < E) atomicAdd(&deg[dst[e]], 1.0f);
}

__global__ void gcnt_kernel(const int* __restrict__ batch,
                            float* __restrict__ gcnt, int N) {
  int n = blockIdx.x * 256 + threadIdx.x;
  if (n < N) atomicAdd(&gcnt[batch[n]], 1.0f);
}

__global__ void asm_initial(const int* __restrict__ atom,
                            const int* __restrict__ ident,
                            const float* __restrict__ emb,
                            const float* __restrict__ embid,
                            bf16* __restrict__ X, int N, int Mpad) {
  int idx = blockIdx.x * 256 + threadIdx.x;
  if (idx >= Mpad * 192) return;
  int n = idx / 192, c = idx % 192;
  float v = 0.0f;
  if (n < N)
    v = (c < 64) ? embid[(size_t)ident[n] * 64 + c]
                 : emb[(size_t)atom[n] * 128 + (c - 64)];
  X[idx] = (bf16)v;
}

__global__ void asm_node(const float* __restrict__ feats,
                         const float* __restrict__ m_sum,
                         const float* __restrict__ deg, bf16* __restrict__ X,
                         int N, int Mpad) {
  int idx = blockIdx.x * 256 + threadIdx.x;
  if (idx >= Mpad * 192) return;
  int n = idx / 192, c = idx % 192;
  float v = 0.0f;
  if (n < N)
    v = (c < 128) ? feats[(size_t)n * 128 + c]
                  : m_sum[(size_t)n * 64 + (c - 128)] / fmaxf(deg[n], 1.0f);
  X[idx] = (bf16)v;
}

__global__ void coords_update(const float* __restrict__ cin,
                              const float* __restrict__ cwrel,
                              const float* __restrict__ deg,
                              float* __restrict__ cout, int N) {
  int idx = blockIdx.x * 256 + threadIdx.x;
  if (idx >= N * 3) return;
  int n = idx / 3;
  cout[idx] = cin[idx] + cwrel[idx] / fmaxf(deg[n], 1.0f);
}

__global__ void asm_fcat(const float* __restrict__ f0,
                         const float* __restrict__ f1,
                         const float* __restrict__ f2, bf16* __restrict__ X,
                         int N, int Mpad) {
  int idx = blockIdx.x * 256 + threadIdx.x;
  if (idx >= Mpad * 384) return;
  int n = idx / 384, c = idx % 384;
  float v = 0.0f;
  if (n < N) {
    const float* f = (c < 128) ? f0 : (c < 256) ? f1 : f2;
    v = silu(f[(size_t)n * 128 + (c & 127)]);
  }
  X[idx] = (bf16)v;
}

__global__ void gsum_kernel(const float* __restrict__ H,
                            const int* __restrict__ batch,
                            float* __restrict__ gsum, int N) {
  int idx = blockIdx.x * 256 + threadIdx.x;
  if (idx >= N * 256) return;
  int n = idx / 256, c = idx % 256;
  atomicAdd(&gsum[(size_t)batch[n] * 256 + c], H[idx]);
}

// Group head: mean -> silu(W1 h+b1) -> silu(W2 t+b2) -> w3.t + b3
__global__ __launch_bounds__(256) void head_kernel(
    const float* __restrict__ gsum, const float* __restrict__ gcnt,
    const float* __restrict__ w1, const float* __restrict__ b1,
    const float* __restrict__ w2, const float* __restrict__ b2,
    const float* __restrict__ w3, const float* __restrict__ b3,
    float* __restrict__ out) {
  __shared__ float h[256], t[256];
  int g = blockIdx.x, c = threadIdx.x;
  float cnt = fmaxf(gcnt[g], 1.0f);
  h[c] = gsum[(size_t)g * 256 + c] / cnt;
  __syncthreads();
  float acc = b1[c];
  for (int k = 0; k < 256; ++k) acc += h[k] * w1[(size_t)k * 256 + c];
  t[c] = silu(acc);
  __syncthreads();
  acc = b2[c];
  for (int k = 0; k < 256; ++k) acc += t[k] * w2[(size_t)k * 256 + c];
  h[c] = silu(acc);
  __syncthreads();
  t[c] = h[c] * w3[c];
  __syncthreads();
  for (int s = 128; s > 0; s >>= 1) {
    if (c < s) t[c] += t[c + s];
    __syncthreads();
  }
  if (c == 0) out[g] = t[0] + b3[0];
}

// ---------------------------------------------------------------------------
// Host-side launch
// ---------------------------------------------------------------------------
// Input order = setup_inputs() dict / pytree insertion order:
// 0 coords | 1 emb | 2 emb_id | 3 initial.w | 4 initial.b |
// 5..16  kernels[0]: e1w,e1b,e2w,e2b,c1w,c1b,c2w,c2b,n1w,n1b,n2w,n2b |
// 17..28 kernels[1] (same) | 29..34 fnn (w,b)x3 | 35..40 fnn2 (w,b)x3 |
// 41 atomids | 42 identity | 43 edge_index | 44 batch
extern "C" void kernel_launch(void* const* d_in, const int* in_sizes, int n_in,
                              void* d_out, int out_size, void* d_ws,
                              size_t ws_size, hipStream_t stream) {
  const int N = in_sizes[41];
  const int E = in_sizes[43] / 2;
  const int G = out_size;
  const int Mpad = (N + 63) & ~63;

  const float* coords0 = (const float*)d_in[0];
  const int* atomids = (const int*)d_in[41];
  const int* identity = (const int*)d_in[42];
  const int* e_src = (const int*)d_in[43];
  const int* e_dst = e_src + E;
  const int* batch = (const int*)d_in[44];

  // bump allocator on d_ws
  char* ws = (char*)d_ws;
  size_t off = 0;
  auto alloc = [&](size_t bytes) -> void* {
    void* p = ws + off;
    off = (off + bytes + 255) & ~(size_t)255;
    return p;
  };

  // ---- packed bf16 weight images (Wt[n][Kpad]) ----
  struct PW { int din, dout, Kpad, Npad; bf16* p; };
  auto pack = [&](int widx, int din, int dout, int Kpad, int Npad) -> bf16* {
    bf16* dst = (bf16*)alloc((size_t)Npad * Kpad * sizeof(bf16));
    int total = Npad * Kpad;
    pack_wt<<<(total + 255) / 256, 256, 0, stream>>>(
        (const float*)d_in[widx], dst, din, dout, Kpad, Npad);
    return dst;
  };
  bf16* wt_init = pack(3, 192, 128, 192, 128);
  bf16* wt_e1[2], * wt_e2[2], * wt_c1[2], * wt_n1[2], * wt_n2[2];
  for (int L = 0; L < 2; ++L) {
    int kb = 5 + L * 12;
    wt_e1[L] = pack(kb + 0, 265, 530, 288, 544);
    wt_e2[L] = pack(kb + 2, 530, 64, 544, 64);
    wt_c1[L] = pack(kb + 4, 64, 256, 64, 256);
    wt_n1[L] = pack(kb + 8, 192, 256, 192, 256);
    wt_n2[L] = pack(kb + 10, 256, 128, 256, 128);
  }
  bf16* wt_f0 = pack(29, 384, 256, 384, 256);
  bf16* wt_f1 = pack(31, 256, 256, 256, 256);
  bf16* wt_f2 = pack(33, 256, 256, 256, 256);
  bf16* wt_g0 = pack(35, 256, 256, 256, 256);
  bf16* wt_g1 = pack(37, 256, 256, 256, 256);
  (void)wt_g0; (void)wt_g1;  // head uses raw f32 weights; images kept for parity

  // ---- activations / accumulators ----
  float* feats[3];
  for (int i = 0; i < 3; ++i)
    feats[i] = (float*)alloc((size_t)Mpad * 128 * sizeof(float));
  float* coordsA = (float*)alloc((size_t)Mpad * 3 * sizeof(float));
  float* coordsB = (float*)alloc((size_t)Mpad * 3 * sizeof(float));
  float* m_sum = (float*)alloc((size_t)Mpad * 64 * sizeof(float));
  float* cwrel = (float*)alloc((size_t)Mpad * 3 * sizeof(float));
  float* deg = (float*)alloc((size_t)Mpad * sizeof(float));
  bf16* Xn = (bf16*)alloc((size_t)Mpad * 192 * sizeof(bf16));
  bf16* Xf = (bf16*)alloc((size_t)Mpad * 384 * sizeof(bf16));
  bf16* Hb1 = (bf16*)alloc((size_t)Mpad * 256 * sizeof(bf16));
  bf16* Hb2 = (bf16*)alloc((size_t)Mpad * 256 * sizeof(bf16));
  float* Hf32 = (float*)alloc((size_t)Mpad * 256 * sizeof(float));
  float* gsum = (float*)alloc((size_t)G * 256 * sizeof(float));
  float* gcnt = (float*)alloc((size_t)G * sizeof(float));
  (void)ws_size;

  const int gemmBlocks = Mpad / 64;
  const int edgeBlocks = (E + 63) / 64;

  // degree (count of edges per dst) — shared by both layers
  hipMemsetAsync(deg, 0, (size_t)Mpad * sizeof(float), stream);
  deg_kernel<<<(E + 255) / 256, 256, 0, stream>>>(e_dst, deg, E);

  // initial embedding: X0 = [emb_id[id], emb[atom]] ; feats0 = silu(X0 W + b)
  asm_initial<<<(Mpad * 192 + 255) / 256, 256, 0, stream>>>(
      atomids, identity, (const float*)d_in[1], (const float*)d_in[2], Xn, N,
      Mpad);
  gemm16<<<gemmBlocks, 128, 0, stream>>>(Xn, wt_init, (const float*)d_in[4], N,
                                         192, 128, feats[0], nullptr, nullptr,
                                         1);

  const float* curC = coords0;
  float* nextC = coordsA;
  for (int L = 0; L < 2; ++L) {
    int kb = 5 + L * 12;
    hipMemsetAsync(m_sum, 0, (size_t)Mpad * 64 * sizeof(float), stream);
    hipMemsetAsync(cwrel, 0, (size_t)Mpad * 3 * sizeof(float), stream);
    edge_kernel<<<edgeBlocks, 128, 0, stream>>>(
        curC, feats[L], e_src, e_dst, wt_e1[L], (const float*)d_in[kb + 1],
        wt_e2[L], (const float*)d_in[kb + 3], wt_c1[L],
        (const float*)d_in[kb + 5], (const float*)d_in[kb + 6],
        (const float*)d_in[kb + 7], m_sum, cwrel, E);
    // node update: feats[L+1] = feats[L] + W2 silu(W1 [feats[L], m_mean] + b1) + b2
    asm_node<<<(Mpad * 192 + 255) / 256, 256, 0, stream>>>(feats[L], m_sum, deg,
                                                           Xn, N, Mpad);
    gemm16<<<gemmBlocks, 128, 0, stream>>>(Xn, wt_n1[L],
                                           (const float*)d_in[kb + 9], N, 192,
                                           256, nullptr, Hb1, nullptr, 1);
    gemm16<<<gemmBlocks, 128, 0, stream>>>(Hb1, wt_n2[L],
                                           (const float*)d_in[kb + 11], N, 256,
                                           128, feats[L + 1], nullptr, feats[L],
                                           0);
    coords_update<<<(N * 3 + 255) / 256, 256, 0, stream>>>(curC, cwrel, deg,
                                                           nextC, N);
    curC = nextC;
    nextC = coordsB;
  }

  // fnn: Xf = silu(concat(f0,f1,f2)) -> 256 -> 256 -> 256 (all silu)
  asm_fcat<<<(Mpad * 384 + 255) / 256, 256, 0, stream>>>(feats[0], feats[1],
                                                         feats[2], Xf, N, Mpad);
  gemm16<<<gemmBlocks, 128, 0, stream>>>(Xf, wt_f0, (const float*)d_in[30], N,
                                         384, 256, nullptr, Hb1, nullptr, 1);
  gemm16<<<gemmBlocks, 128, 0, stream>>>(Hb1, wt_f1, (const float*)d_in[32], N,
                                         256, 256, nullptr, Hb2, nullptr, 1);
  gemm16<<<gemmBlocks, 128, 0, stream>>>(Hb2, wt_f2, (const float*)d_in[34], N,
                                         256, 256, Hf32, nullptr, nullptr, 1);

  // group mean + head
  hipMemsetAsync(gsum, 0, (size_t)G * 256 * sizeof(float), stream);
  hipMemsetAsync(gcnt, 0, (size_t)G * sizeof(float), stream);
  gcnt_kernel<<<(N + 255) / 256, 256, 0, stream>>>(batch, gcnt, N);
  gsum_kernel<<<(N * 256 + 255) / 256, 256, 0, stream>>>(Hf32, batch, gsum, N);
  head_kernel<<<G, 256, 0, stream>>>(
      gsum, gcnt, (const float*)d_in[35], (const float*)d_in[36],
      (const float*)d_in[37], (const float*)d_in[38], (const float*)d_in[39],
      (const float*)d_in[40], (float*)d_out);
}
